// Attention_35940286333182
// MI455X (gfx1250) — compile-verified, matrix operand order checked by reading, not run on hardware
//
#include <hip/hip_runtime.h>
#include <math.h>

// ---------------------------------------------------------------------------
// Types for CDNA5 WMMA (wave32): v16bf A/B fragments, v8f C/D accumulators.
// ---------------------------------------------------------------------------
typedef __attribute__((ext_vector_type(16))) __bf16 v16bf;
typedef __attribute__((ext_vector_type(8)))  float  v8f;

static __device__ __forceinline__ v8f wmma_bf16(v16bf a, v16bf b, v8f c) {
    // D = A(16x32 bf16) * B(32x16 bf16) + C(16x16 f32)
    return __builtin_amdgcn_wmma_f32_16x16x32_bf16(
        /*neg_a=*/false, a, /*neg_b=*/false, b,
        /*c_mod=*/(short)0, c, /*reuse_a=*/false, /*reuse_b=*/false);
}

// Load one 16-element bf16 fragment for the 16x16x32 WMMA A/B layout:
//   lane m = lane&15 selects the row; half = lane>>4 selects the K sub-chunks:
//   elements 0..7  <- k = 8*half .. 8*half+7
//   elements 8..15 <- k = 16+8*half .. 16+8*half+7
// 'p' points at (row, k=0) of a row-major tile; both chunks are 16B loads.
static __device__ __forceinline__ v16bf ldfrag(const __bf16* p, int half) {
    union { v16bf v; uint4 q[2]; } f;
    f.q[0] = *(const uint4*)(p + 8 * half);
    f.q[1] = *(const uint4*)(p + 8 * half + 16);
    return f.v;
}

// ---------------------------------------------------------------------------
// CDNA5 async global->LDS copy (ASYNCcnt path): each active lane copies 16B
// from its global address to its LDS byte offset, no VGPR data movement.
//   LDS address comes from the vdst VGPR; tracked with ASYNCcnt.
// The low 32 bits of a generic pointer to __shared__ are the LDS offset.
// ---------------------------------------------------------------------------
static __device__ __forceinline__ void async_copy_b128(void* lds_dst,
                                                       const void* gsrc) {
    unsigned           loff  = (unsigned)(unsigned long long)lds_dst;
    unsigned long long gaddr = (unsigned long long)gsrc;
    asm volatile("global_load_async_to_lds_b128 %0, %1, off"
                 :: "v"(loff), "v"(gaddr)
                 : "memory");
}
static __device__ __forceinline__ void wait_async_all() {
    asm volatile("s_wait_asynccnt 0x0" ::: "memory");
}

// ---------------------------------------------------------------------------
// Tiled GEMM: C[M,N] = A[M,K] x B[K,N]
//   Block: 256 threads (8 waves), tile 128(M) x 128(N) x 32(K).
//   Wave grid 4x2: each wave computes 32x64 = 2x4 WMMA tiles.
//   A: fp32 (converted to bf16 on LDS store) or bf16 (async copied to LDS).
//   B: fp32, converted + transposed into LDS as Bs[n][k] so B-fragments are
//      contiguous along k (same pattern as A-fragments).
// Epilogue:
//   HEAD_OUT=true : write bf16 to [B, H, Nseq, 64] layout, scaled (Q gets 1/8).
//   HEAD_OUT=false: write fp32 row-major with bias add.
// ---------------------------------------------------------------------------
template <bool A_BF16, bool HEAD_OUT>
__global__ __launch_bounds__(256) void gemm_wmma_kernel(
    const void* __restrict__ Av, const float* __restrict__ Bp,
    void* __restrict__ Cv, const float* __restrict__ bias,
    int M, int N, int K, int Nseq, float out_scale)
{
    __shared__ __bf16 As[128][40];   // [m][k], row stride 80B (16B multiple)
    __shared__ __bf16 Bs[128][40];   // [n][k], transposed on store

    const int tid  = threadIdx.x;
    const int lane = tid & 31;
    const int wave = tid >> 5;
    const int wm   = wave >> 1;       // 0..3
    const int wn   = wave & 1;        // 0..1
    const int half = lane >> 4;
    const int lm   = lane & 15;

    v8f acc[2][4];
    for (int im = 0; im < 2; ++im)
        for (int jn = 0; jn < 4; ++jn)
            acc[im][jn] = (v8f){};

    for (int k0 = 0; k0 < K; k0 += 32) {
        bool used_async = false;
        // ---- stage A tile (128x32) ----
        if (A_BF16) {
            const __bf16* A = (const __bf16*)Av;
            #pragma unroll
            for (int i = 0; i < 2; ++i) {
                int idx = tid + i * 256;          // 512 chunks of 8 bf16
                int row = idx >> 2, c8 = idx & 3;
                const __bf16* src =
                    A + (size_t)(blockIdx.y * 128 + row) * K + k0 + c8 * 8;
                async_copy_b128(&As[row][c8 * 8], src);   // ASYNCcnt path
            }
            used_async = true;
        } else {
            const float* A = (const float*)Av;
            #pragma unroll
            for (int i = 0; i < 4; ++i) {
                int idx = tid + i * 256;          // 1024 float4 chunks
                int row = idx >> 3, c4 = idx & 7;
                float4 f = *(const float4*)(
                    A + (size_t)(blockIdx.y * 128 + row) * K + k0 + c4 * 4);
                union { __bf16 h[4]; uint2 u; } pk;
                pk.h[0] = (__bf16)f.x; pk.h[1] = (__bf16)f.y;
                pk.h[2] = (__bf16)f.z; pk.h[3] = (__bf16)f.w;
                *(uint2*)(&As[row][c4 * 4]) = pk.u;
            }
        }
        // ---- stage B tile (32x128), transposed into Bs[n][k] ----
        #pragma unroll
        for (int i = 0; i < 4; ++i) {
            int idx = tid + i * 256;              // 1024 float4 chunks
            int kr = idx >> 5, c4 = idx & 31;
            float4 f = *(const float4*)(
                Bp + (size_t)(k0 + kr) * N + blockIdx.x * 128 + c4 * 4);
            Bs[c4 * 4 + 0][kr] = (__bf16)f.x;
            Bs[c4 * 4 + 1][kr] = (__bf16)f.y;
            Bs[c4 * 4 + 2][kr] = (__bf16)f.z;
            Bs[c4 * 4 + 3][kr] = (__bf16)f.w;
        }
        // ---- prefetch next k-tile into cache while we compute this one ----
        if (k0 + 32 < K) {
            if (A_BF16) {
                const __bf16* A = (const __bf16*)Av;
                __builtin_prefetch(
                    A + (size_t)(blockIdx.y * 128 + (tid >> 1)) * K +
                        (k0 + 32) + (tid & 1) * 16, 0, 0);
            } else {
                const float* A = (const float*)Av;
                __builtin_prefetch(
                    A + (size_t)(blockIdx.y * 128 + (tid >> 1)) * K +
                        (k0 + 32) + (tid & 1) * 16, 0, 0);
            }
            __builtin_prefetch(
                Bp + (size_t)(k0 + 32 + (tid >> 3)) * N +
                    blockIdx.x * 128 + (tid & 7) * 16, 0, 0);
        }
        if (used_async) wait_async_all();
        __syncthreads();

        v16bf afrag[2], bfrag[4];
        #pragma unroll
        for (int im = 0; im < 2; ++im)
            afrag[im] = ldfrag(&As[wm * 32 + im * 16 + lm][0], half);
        #pragma unroll
        for (int jn = 0; jn < 4; ++jn)
            bfrag[jn] = ldfrag(&Bs[wn * 64 + jn * 16 + lm][0], half);

        #pragma unroll
        for (int im = 0; im < 2; ++im)
            #pragma unroll
            for (int jn = 0; jn < 4; ++jn)
                acc[im][jn] = wmma_bf16(afrag[im], bfrag[jn], acc[im][jn]);

        __syncthreads();
    }

    // ---- epilogue ----
    #pragma unroll
    for (int im = 0; im < 2; ++im) {
        #pragma unroll
        for (int jn = 0; jn < 4; ++jn) {
            #pragma unroll
            for (int j = 0; j < 8; ++j) {
                int m = blockIdx.y * 128 + wm * 32 + im * 16 + j + 8 * half;
                int n = blockIdx.x * 128 + wn * 64 + jn * 16 + lm;
                float v = acc[im][jn][j];
                if (HEAD_OUT) {
                    // [B, H, Nseq, 64] bf16
                    int bb = m / Nseq, nr = m % Nseq;
                    int hh = n >> 6,  dd = n & 63;
                    ((__bf16*)Cv)[(((size_t)bb * 16 + hh) * Nseq + nr) * 64 + dd]
                        = (__bf16)(v * out_scale);
                } else {
                    float bv = bias ? bias[n] : 0.0f;
                    ((float*)Cv)[(size_t)m * N + n] = v + bv;
                }
            }
        }
    }
}

// ---------------------------------------------------------------------------
// Flash attention: one block per (batch, head, 64-query-row tile).
//   128 threads (4 waves); wave w owns query rows [w*16, w*16+16).
//   Loop over 64-key tiles: K tile staged with async global->LDS copies
//   (no conversion needed), V tile transposed through VGPRs.
//   S = Q*K^T (WMMA), online softmax via LDS, O = O*rescale + P*V (WMMA).
//   Q is pre-scaled by 1/sqrt(d).
// ---------------------------------------------------------------------------
__global__ __launch_bounds__(128) void attn_wmma_kernel(
    const __bf16* __restrict__ Q, const __bf16* __restrict__ Kk,
    const __bf16* __restrict__ V, __bf16* __restrict__ O)
{
    const int NSEQ = 2048, D = 64, H = 16;
    const int q0   = blockIdx.x * 64;
    const int h    = blockIdx.y;
    const int b    = blockIdx.z;
    const int tid  = threadIdx.x;
    const int lane = tid & 31;
    const int wave = tid >> 5;
    const int half = lane >> 4;
    const int lm   = lane & 15;

    const size_t headoff = ((size_t)(b * H + h)) * NSEQ * D;
    const __bf16* Qh = Q + headoff;
    const __bf16* Kh = Kk + headoff;
    const __bf16* Vh = V + headoff;

    __shared__ __bf16 Ks[64][72];     // [key][d]  (row-major -> K^T B-frags)
    __shared__ __bf16 Vs[64][72];     // [d][key]  (transposed -> V B-frags)
    __shared__ float  Sp[64][72];     // scores fp32
    __shared__ __bf16 Ps[64][72];     // softmax probs bf16
    __shared__ float  rowscale[64];
    __shared__ float  rowinv[64];

    // Q fragments: wave's 16 rows, k = d in [0,64) -> 2 fragments.
    v16bf qf[2];
    {
        const __bf16* qrow = Qh + (size_t)(q0 + wave * 16 + lm) * D;
        qf[0] = ldfrag(qrow, half);
        qf[1] = ldfrag(qrow + 32, half);
    }

    v8f of[4];
    #pragma unroll
    for (int jt = 0; jt < 4; ++jt) of[jt] = (v8f){};
    float m_run = -1e30f, l_run = 0.0f;   // per-row state in threads 0..63

    for (int kt = 0; kt < NSEQ / 64; ++kt) {
        const int kbase = kt * 64;
        // ---- K tile: pure bf16 copy -> async global->LDS (ASYNCcnt) ----
        #pragma unroll
        for (int i = 0; i < 4; ++i) {
            int idx = tid + i * 128;           // 512 chunks of 8 bf16
            int row = idx >> 2, c8 = idx & 3;
            async_copy_b128(&Ks[row][c8 * 8],
                            Kh + (size_t)(kbase + row) * D + c8 * 8);
        }
        // ---- V tile: needs transpose, stage through VGPRs ----
        #pragma unroll
        for (int i = 0; i < 4; ++i) {
            int idx = tid + i * 128;
            int row = idx >> 2, c8 = idx & 3;
            union { uint4 u; __bf16 e[8]; } t;
            t.u = *(const uint4*)(Vh + (size_t)(kbase + row) * D + c8 * 8);
            #pragma unroll
            for (int e = 0; e < 8; ++e) Vs[c8 * 8 + e][row] = t.e[e];
        }
        // ---- prefetch next K/V tiles ----
        if (kt + 1 < NSEQ / 64) {
            const __bf16* nk =
                Kh + (size_t)(kbase + 64 + (tid >> 1)) * D + (tid & 1) * 32;
            const __bf16* nv =
                Vh + (size_t)(kbase + 64 + (tid >> 1)) * D + (tid & 1) * 32;
            __builtin_prefetch(nk, 0, 0);
            __builtin_prefetch(nv, 0, 0);
        }
        wait_async_all();
        __syncthreads();

        // ---- S = Q * K^T for this wave's 16-row strip (4 col tiles) ----
        #pragma unroll
        for (int jt = 0; jt < 4; ++jt) {
            v16bf kf0 = ldfrag(&Ks[jt * 16 + lm][0], half);
            v16bf kf1 = ldfrag(&Ks[jt * 16 + lm][32], half);
            v8f s = (v8f){};
            s = wmma_bf16(qf[0], kf0, s);
            s = wmma_bf16(qf[1], kf1, s);
            #pragma unroll
            for (int j = 0; j < 8; ++j)
                Sp[wave * 16 + j + 8 * half][jt * 16 + lm] = s[j];
        }
        __syncthreads();

        // ---- online softmax: thread t owns row t (t < 64) ----
        if (tid < 64) {
            float mx = m_run;
            #pragma unroll 8
            for (int j = 0; j < 64; ++j) mx = fmaxf(mx, Sp[tid][j]);
            float r = __expf(m_run - mx);
            float sum = 0.0f;
            #pragma unroll 8
            for (int j = 0; j < 64; ++j) {
                float p = __expf(Sp[tid][j] - mx);
                sum += p;
                Ps[tid][j] = (__bf16)p;
            }
            l_run = l_run * r + sum;
            m_run = mx;
            rowscale[tid] = r;
        }
        __syncthreads();

        // ---- O = O * rescale + P * V ----
        float sc[8];
        #pragma unroll
        for (int j = 0; j < 8; ++j)
            sc[j] = rowscale[wave * 16 + j + 8 * half];
        #pragma unroll
        for (int jt = 0; jt < 4; ++jt)
            #pragma unroll
            for (int j = 0; j < 8; ++j) of[jt][j] *= sc[j];

        #pragma unroll
        for (int ks = 0; ks < 2; ++ks) {
            v16bf pf = ldfrag(&Ps[wave * 16 + lm][ks * 32], half);
            #pragma unroll
            for (int jt = 0; jt < 4; ++jt) {
                v16bf vf = ldfrag(&Vs[jt * 16 + lm][ks * 32], half);
                of[jt] = wmma_bf16(pf, vf, of[jt]);
            }
        }
        __syncthreads();
    }

    if (tid < 64) rowinv[tid] = 1.0f / l_run;
    __syncthreads();

    float inv[8];
    #pragma unroll
    for (int j = 0; j < 8; ++j)
        inv[j] = rowinv[wave * 16 + j + 8 * half];

    // store O as bf16 in [B, Nseq, H*64] (row-major for the output GEMM)
    #pragma unroll
    for (int jt = 0; jt < 4; ++jt) {
        #pragma unroll
        for (int j = 0; j < 8; ++j) {
            int qr  = q0 + wave * 16 + j + 8 * half;
            int col = h * 64 + jt * 16 + lm;
            O[((size_t)(b * NSEQ) + qr) * (H * D) + col] =
                (__bf16)(of[jt][j] * inv[j]);
        }
    }
}

// ---------------------------------------------------------------------------
// Launch: Q/K/V projections (fp32 in -> bf16 per-head ws), flash attention,
// output projection (bf16 ws -> fp32 out + bias).
// ---------------------------------------------------------------------------
extern "C" void kernel_launch(void* const* d_in, const int* in_sizes, int n_in,
                              void* d_out, int out_size, void* d_ws, size_t ws_size,
                              hipStream_t stream) {
    (void)in_sizes; (void)n_in; (void)out_size; (void)ws_size;

    const float* q_in = (const float*)d_in[0];
    const float* k_in = (const float*)d_in[1];
    const float* v_in = (const float*)d_in[2];
    const float* Wq   = (const float*)d_in[3];
    const float* Wk   = (const float*)d_in[4];
    const float* Wv   = (const float*)d_in[5];
    const float* Wo   = (const float*)d_in[6];
    const float* bo   = (const float*)d_in[7];
    float*       out  = (float*)d_out;

    const int B = 2, NSEQ = 2048, DM = 1024, H = 16, DH = 64;
    const int M = B * NSEQ;             // 4096
    const float SCALE = 0.125f;         // 1/sqrt(64), exact in bf16

    const size_t elems = (size_t)B * NSEQ * DM;   // 4M elements per tensor
    __bf16* q_ws = (__bf16*)d_ws;
    __bf16* k_ws = q_ws + elems;
    __bf16* v_ws = k_ws + elems;
    __bf16* o_ws = v_ws + elems;

    dim3 gblk(256);
    dim3 ggrid(DM / 128, M / 128);      // (8, 32)

    // Projections: fp32 activations/weights -> bf16 [B,H,N,64] (Q scaled).
    gemm_wmma_kernel<false, true><<<ggrid, gblk, 0, stream>>>(
        q_in, Wq, q_ws, nullptr, M, DM, DM, NSEQ, SCALE);
    gemm_wmma_kernel<false, true><<<ggrid, gblk, 0, stream>>>(
        k_in, Wk, k_ws, nullptr, M, DM, DM, NSEQ, 1.0f);
    gemm_wmma_kernel<false, true><<<ggrid, gblk, 0, stream>>>(
        v_in, Wv, v_ws, nullptr, M, DM, DM, NSEQ, 1.0f);

    // Flash attention per (q-tile, head, batch).
    dim3 agrid(NSEQ / 64, H, B);        // (32, 16, 2)
    attn_wmma_kernel<<<agrid, dim3(128), 0, stream>>>(q_ws, k_ws, v_ws, o_ws);

    // Output projection: bf16 O x fp32 Wo + bias -> fp32 out.
    gemm_wmma_kernel<true, false><<<ggrid, gblk, 0, stream>>>(
        o_ws, Wo, out, bo, M, DM, (int)(H * DH), NSEQ, 1.0f);
}
